// Llama4MegaBlocksAdapter_1460288880659
// MI455X (gfx1250) — matrix operand order repaired and checked by reading, not compiled, not run on hardware
//
#include <hip/hip_runtime.h>

// ---------------------------------------------------------------------------
// Llama4-style top-1 MoE GLU for MI455X (gfx1250).
// bf16 WMMA (v_wmma_f32_16x16x32_bf16) + double-buffered LDS with
// GLOBAL_LOAD_ASYNC_TO_LDS_B128 staging of the bf16 activation tiles.
// T=8192 tokens, H=I=2048, E=8 experts, top-1 routing + shared expert.
// ---------------------------------------------------------------------------

#define Tn 8192
#define Hn 2048
#define In 2048
#define En 8

#define BM 128            // token rows per block tile
#define BN 64             // output cols per block tile
#define BK 32             // K step == WMMA K
#define LDA (BK + 8)      // LDS row stride (ushorts) for A tile
#define LDB (BN + 8)      // LDS row stride (ushorts) for B tiles

typedef __bf16 v16bf __attribute__((ext_vector_type(16)));
typedef float  v8f   __attribute__((ext_vector_type(8)));

__device__ __forceinline__ unsigned short f2bf(float f) {
    unsigned u = __builtin_bit_cast(unsigned, f);
    u += 0x7fffu + ((u >> 16) & 1u);          // round-to-nearest-even
    return (unsigned short)(u >> 16);
}

// Async copy 32B (one A-tile half-row) global bf16 -> LDS. ASYNCcnt-tracked.
// IOFFSET applies to both LDS and global sides (CDNA5 ISA).
__device__ __forceinline__ void async_copy32(const unsigned short* gsrc,
                                             unsigned short* lds_dst) {
    unsigned lds = (unsigned)(unsigned long long)lds_dst;
    unsigned long long ga = (unsigned long long)gsrc;
    asm volatile("global_load_async_to_lds_b128 %0, %1, off\n\t"
                 "global_load_async_to_lds_b128 %0, %1, off offset:16"
                 :: "v"(lds), "v"(ga) : "memory");
}
__device__ __forceinline__ void wait_async0() {
    asm volatile("s_wait_asynccnt 0x0" ::: "memory");
}

// --- WMMA fragment builders from LDS tiles (layouts per CDNA5 ISA)
__device__ __forceinline__ v16bf frag_a(const unsigned short* As, int m0) {
    const int lane = threadIdx.x & 31;
    const int half = lane >> 4;
    const unsigned short* row = As + (m0 + (lane & 15)) * LDA;
    v16bf a;
#pragma unroll
    for (int j = 0; j < 16; ++j) {
        int k = (j < 8) ? (half * 8 + j) : (8 + half * 8 + j); // j>=8: 16+half*8+(j-8)
        a[j] = __builtin_bit_cast(__bf16, row[k]);
    }
    return a;
}

__device__ __forceinline__ v16bf frag_b(const unsigned short* Bs, int n0) {
    const int lane = threadIdx.x & 31;
    const int half = lane >> 4;
    const unsigned short* col = Bs + n0 + (lane & 15);
    v16bf b;
#pragma unroll
    for (int j = 0; j < 16; ++j) {
        int k = (j < 8) ? (half * 8 + j) : (8 + half * 8 + j);
        b[j] = __builtin_bit_cast(__bf16, col[k * LDB]);
    }
    return b;
}

// ---------------------------------------------------------------------------
// Small prep kernels
// ---------------------------------------------------------------------------
__global__ void __launch_bounds__(256) zero_counts_kernel(int* counts) {
    if (threadIdx.x < En) counts[threadIdx.x] = 0;
}

__global__ void __launch_bounds__(256) convert_x_kernel(const float* __restrict__ x,
                                                        unsigned short* __restrict__ xbf) {
    size_t i = ((size_t)blockIdx.x * 256 + threadIdx.x) * 8;
    const float4* p = (const float4*)(x + i);
    float4 a = p[0], b = p[1];
    unsigned short t[8] = { f2bf(a.x), f2bf(a.y), f2bf(a.z), f2bf(a.w),
                            f2bf(b.x), f2bf(b.y), f2bf(b.z), f2bf(b.w) };
    *(uint4*)(xbf + i) = *(uint4*)t;
}

__global__ void __launch_bounds__(256) router_kernel(const float* __restrict__ x,
                                                     const float* __restrict__ rw,
                                                     int* __restrict__ counts,
                                                     int* __restrict__ eidx,
                                                     int* __restrict__ eslot,
                                                     float* __restrict__ ewt,
                                                     int* __restrict__ tlist) {
    int t    = (int)((blockIdx.x * 256 + threadIdx.x) >> 5);
    int lane = threadIdx.x & 31;
    if (t >= Tn) return;
    const float* xr = x + (size_t)t * Hn;
    float acc[En];
#pragma unroll
    for (int e = 0; e < En; ++e) acc[e] = 0.f;
    for (int h = lane; h < Hn; h += 32) {
        float xv = xr[h];
#pragma unroll
        for (int e = 0; e < En; ++e) acc[e] += xv * rw[e * Hn + h];
    }
#pragma unroll
    for (int e = 0; e < En; ++e) {
#pragma unroll
        for (int off = 16; off > 0; off >>= 1) acc[e] += __shfl_down(acc[e], off, 32);
    }
    if (lane == 0) {
        int best = 0; float bv = acc[0];
#pragma unroll
        for (int e = 1; e < En; ++e)
            if (acc[e] > bv) { bv = acc[e]; best = e; }
        eidx[t] = best;
        ewt[t]  = 1.f / (1.f + __expf(-bv));
        int s   = atomicAdd(&counts[best], 1);
        eslot[t] = s;
        tlist[best * Tn + s] = t;
    }
}

__global__ void prefix_kernel(const int* counts, int* base) {
    if (threadIdx.x == 0 && blockIdx.x == 0) {
        int s = 0;
        for (int e = 0; e < En; ++e) { base[e] = s; s += counts[e]; }
    }
}

__global__ void __launch_bounds__(256) build_slots_kernel(const int* __restrict__ eidx,
                                                          const int* __restrict__ eslot,
                                                          const float* __restrict__ ewt,
                                                          const int* __restrict__ base,
                                                          int* __restrict__ tok_of_slot,
                                                          float* __restrict__ wslot) {
    int t = blockIdx.x * 256 + threadIdx.x;
    if (t >= Tn) return;
    int slot = base[eidx[t]] + eslot[t];
    tok_of_slot[slot] = t;
    wslot[slot]       = ewt[t];
}

// ---------------------------------------------------------------------------
// GLU stage 1: h = silu(x@Wg) * (x@Wu). Double-buffered LDS; A via async copy.
// EXPERT=true : gathered token rows, Wg/Wu row-major [H,I] per expert.
// EXPERT=false: dense rows, Wg/Wu are [I,H] used transposed.
// ---------------------------------------------------------------------------
template <bool EXPERT>
__global__ void __launch_bounds__(256) glu_stage1(const unsigned short* __restrict__ xbf,
                                                  const float* __restrict__ Wg_all,
                                                  const float* __restrict__ Wu_all,
                                                  const int* __restrict__ counts,
                                                  const int* __restrict__ base,
                                                  const int* __restrict__ tlist,
                                                  unsigned short* __restrict__ hbuf) {
    const int nb  = blockIdx.x;
    const int e   = EXPERT ? blockIdx.y : 0;
    const int zt  = EXPERT ? blockIdx.z : blockIdx.y;
    int cnt;
    if (EXPERT) cnt = counts[e]; else cnt = Tn;
    const int row0 = zt * BM;
    if (row0 >= cnt) return;                          // block-uniform exit
    const float* Wg = EXPERT ? Wg_all + (size_t)e * Hn * In : Wg_all;
    const float* Wu = EXPERT ? Wu_all + (size_t)e * Hn * In : Wu_all;
    const int i0      = nb * BN;
    const int outbase = EXPERT ? base[e] : 0;

    __shared__ unsigned short As[2][BM * LDA];
    __shared__ unsigned short Bg[2][BK * LDB];
    __shared__ unsigned short Bu[2][BK * LDB];

    const int tid   = threadIdx.x;
    const int arow  = tid >> 1;
    const int ahalf = tid & 1;
    const int tok   = row0 + arow;
    const bool avalid = tok < cnt;
    // invalid rows copy a valid dummy row; their outputs are masked in the
    // epilogue (WMMA row m depends only on A row m), keeping EXEC uniform.
    const unsigned short* asrc;
    if (EXPERT) asrc = xbf + (size_t)(avalid ? tlist[e * Tn + tok] : 0) * Hn;
    else        asrc = xbf + (size_t)tok * Hn;

    float rg[8], ru[8];                               // fp32 B-tile prefetch regs

    auto load_b_regs = [&](int k0) {
        if (EXPERT) {                                  // B[k][n] = W[(k0+k)*In + i0+n]
            int k = tid >> 3, n = (tid & 7) * 8;
            const float* pg = Wg + (size_t)(k0 + k) * In + i0 + n;
            const float* pu = Wu + (size_t)(k0 + k) * In + i0 + n;
            __builtin_prefetch(pg + (size_t)BK * In, 0, 1);
            __builtin_prefetch(pu + (size_t)BK * In, 0, 1);
#pragma unroll
            for (int j = 0; j < 8; ++j) { rg[j] = pg[j]; ru[j] = pu[j]; }
        } else {                                       // B[k][n] = W[(i0+n)*Hn + k0+k]
            int n = tid >> 2, ks = (tid & 3) * 8;
            const float* pg = Wg + (size_t)(i0 + n) * Hn + k0 + ks;
            const float* pu = Wu + (size_t)(i0 + n) * Hn + k0 + ks;
            __builtin_prefetch(pg + BK, 0, 1);
            __builtin_prefetch(pu + BK, 0, 1);
#pragma unroll
            for (int j = 0; j < 8; ++j) { rg[j] = pg[j]; ru[j] = pu[j]; }
        }
    };
    auto store_b_lds = [&](int buf) {
        if (EXPERT) {
            int k = tid >> 3, n = (tid & 7) * 8;
            unsigned short tg[8], tu[8];
#pragma unroll
            for (int j = 0; j < 8; ++j) { tg[j] = f2bf(rg[j]); tu[j] = f2bf(ru[j]); }
            *(uint4*)&Bg[buf][k * LDB + n] = *(uint4*)tg;
            *(uint4*)&Bu[buf][k * LDB + n] = *(uint4*)tu;
        } else {
            int n = tid >> 2, ks = (tid & 3) * 8;
#pragma unroll
            for (int j = 0; j < 8; ++j) {
                Bg[buf][(ks + j) * LDB + n] = f2bf(rg[j]);
                Bu[buf][(ks + j) * LDB + n] = f2bf(ru[j]);
            }
        }
    };
    auto async_a = [&](int k0, int buf) {
        async_copy32(asrc + k0 + ahalf * 16, &As[buf][arow * LDA + ahalf * 16]);
    };

    v8f accg[4], accu[4];
#pragma unroll
    for (int q = 0; q < 4; ++q)
#pragma unroll
        for (int j = 0; j < 8; ++j) { accg[q][j] = 0.f; accu[q][j] = 0.f; }

    const int wv = tid >> 5;
    const int m0 = wv * 16;

    // prologue: tile 0
    load_b_regs(0);
    async_a(0, 0);
    store_b_lds(0);
    wait_async0();
    __syncthreads();

    for (int k0 = 0; k0 < Hn; k0 += BK) {
        const int  buf = (k0 >> 5) & 1;
        const bool hn  = (k0 + BK) < Hn;
        if (hn) { load_b_regs(k0 + BK); async_a(k0 + BK, buf ^ 1); }

        // hoist fragment loads, back-to-back WMMA chain (loads of the next
        // fragment overlap the XDL pipe of the previous WMMA)
        v16bf a = frag_a(&As[buf][0], m0);
        v16bf fb[8];
#pragma unroll
        for (int q = 0; q < 4; ++q) {
            fb[q]     = frag_b(&Bg[buf][0], q * 16);
            fb[4 + q] = frag_b(&Bu[buf][0], q * 16);
        }
#pragma unroll
        for (int q = 0; q < 4; ++q) {
            accg[q] = __builtin_amdgcn_wmma_f32_16x16x32_bf16(false, a, false, fb[q],
                                                              (short)0, accg[q], false, false);
            accu[q] = __builtin_amdgcn_wmma_f32_16x16x32_bf16(false, a, false, fb[4 + q],
                                                              (short)0, accu[q], false, false);
        }

        if (hn) store_b_lds(buf ^ 1);
        wait_async0();
        __syncthreads();
    }

    // epilogue: h = silu(g)*u -> bf16 grouped intermediate
    const int lane = tid & 31;
    const int half = lane >> 4;
    const int ncol = lane & 15;
#pragma unroll
    for (int v = 0; v < 8; ++v) {
        int gr = row0 + m0 + half * 8 + v;
        if (gr < cnt) {
            unsigned short* hrow = hbuf + (size_t)(outbase + gr) * In + i0 + ncol;
#pragma unroll
            for (int q = 0; q < 4; ++q) {
                float g = accg[q][v], u = accu[q][v];
                float hval = (g / (1.f + __expf(-g))) * u;
                hrow[q * 16] = f2bf(hval);
            }
        }
    }
}

// ---------------------------------------------------------------------------
// GLU stage 2: y = h @ Wd. Double-buffered LDS; A (bf16 h) via async copy.
// EXPERT=true : Wd row-major [I,H] per expert; out[t] += weight * y.
// EXPERT=false: Wd is shared_down [H,I] used transposed; out[t] = y.
// ---------------------------------------------------------------------------
template <bool EXPERT>
__global__ void __launch_bounds__(256) glu_stage2(const unsigned short* __restrict__ hbuf,
                                                  const float* __restrict__ Wd_all,
                                                  const int* __restrict__ counts,
                                                  const int* __restrict__ base,
                                                  const int* __restrict__ tok_of_slot,
                                                  const float* __restrict__ wslot,
                                                  float* __restrict__ out) {
    const int nb = blockIdx.x;
    const int e  = EXPERT ? blockIdx.y : 0;
    const int zt = EXPERT ? blockIdx.z : blockIdx.y;
    int cnt;
    if (EXPERT) cnt = counts[e]; else cnt = Tn;
    const int row0 = zt * BM;
    if (row0 >= cnt) return;
    const float* Wd = EXPERT ? Wd_all + (size_t)e * In * Hn : Wd_all;
    const int sbase = EXPERT ? base[e] : 0;
    const int n0    = nb * BN;

    __shared__ unsigned short As[2][BM * LDA];
    __shared__ unsigned short Bs[2][BK * LDB];

    const int tid   = threadIdx.x;
    const int arow  = tid >> 1;
    const int ahalf = tid & 1;
    const int r     = row0 + arow;
    const bool avalid = r < cnt;
    const unsigned short* asrc = hbuf + (size_t)(avalid ? (sbase + r) : 0) * In;

    float rd[8];

    auto load_b_regs = [&](int k0) {
        if (EXPERT) {                                  // B[k][n] = Wd[(k0+k)*Hn + n0+n]
            int k = tid >> 3, n = (tid & 7) * 8;
            const float* pd = Wd + (size_t)(k0 + k) * Hn + n0 + n;
            __builtin_prefetch(pd + (size_t)BK * Hn, 0, 1);
#pragma unroll
            for (int j = 0; j < 8; ++j) rd[j] = pd[j];
        } else {                                       // B[k][n] = Wd[(n0+n)*In + k0+k]
            int n = tid >> 2, ks = (tid & 3) * 8;
            const float* pd = Wd + (size_t)(n0 + n) * In + k0 + ks;
            __builtin_prefetch(pd + BK, 0, 1);
#pragma unroll
            for (int j = 0; j < 8; ++j) rd[j] = pd[j];
        }
    };
    auto store_b_lds = [&](int buf) {
        if (EXPERT) {
            int k = tid >> 3, n = (tid & 7) * 8;
            unsigned short td[8];
#pragma unroll
            for (int j = 0; j < 8; ++j) td[j] = f2bf(rd[j]);
            *(uint4*)&Bs[buf][k * LDB + n] = *(uint4*)td;
        } else {
            int n = tid >> 2, ks = (tid & 3) * 8;
#pragma unroll
            for (int j = 0; j < 8; ++j) Bs[buf][(ks + j) * LDB + n] = f2bf(rd[j]);
        }
    };
    auto async_a = [&](int k0, int buf) {
        async_copy32(asrc + k0 + ahalf * 16, &As[buf][arow * LDA + ahalf * 16]);
    };

    v8f acc[4];
#pragma unroll
    for (int q = 0; q < 4; ++q)
#pragma unroll
        for (int j = 0; j < 8; ++j) acc[q][j] = 0.f;

    const int wv = tid >> 5;
    const int m0 = wv * 16;

    load_b_regs(0);
    async_a(0, 0);
    store_b_lds(0);
    wait_async0();
    __syncthreads();

    for (int k0 = 0; k0 < In; k0 += BK) {
        const int  buf = (k0 >> 5) & 1;
        const bool hn  = (k0 + BK) < In;
        if (hn) { load_b_regs(k0 + BK); async_a(k0 + BK, buf ^ 1); }

        v16bf a = frag_a(&As[buf][0], m0);
        v16bf fb[4];
#pragma unroll
        for (int q = 0; q < 4; ++q) fb[q] = frag_b(&Bs[buf][0], q * 16);
#pragma unroll
        for (int q = 0; q < 4; ++q)
            acc[q] = __builtin_amdgcn_wmma_f32_16x16x32_bf16(false, a, false, fb[q],
                                                             (short)0, acc[q], false, false);

        if (hn) store_b_lds(buf ^ 1);
        wait_async0();
        __syncthreads();
    }

    const int lane = tid & 31;
    const int half = lane >> 4;
    const int ncol = lane & 15;
#pragma unroll
    for (int v = 0; v < 8; ++v) {
        int gr = row0 + m0 + half * 8 + v;
        if (gr < cnt) {
            if (EXPERT) {
                int   slot = sbase + gr;
                int   t    = tok_of_slot[slot];
                float wgt  = wslot[slot];
                float* orow = out + (size_t)t * Hn + n0 + ncol;
#pragma unroll
                for (int q = 0; q < 4; ++q)
                    orow[q * 16] += wgt * acc[q][v];   // exclusive per token
            } else {
                float* orow = out + (size_t)gr * Hn + n0 + ncol;
#pragma unroll
                for (int q = 0; q < 4; ++q)
                    orow[q * 16] = acc[q][v];          // runs first, plain store
            }
        }
    }
}

// ---------------------------------------------------------------------------
// Host launcher
// ---------------------------------------------------------------------------
extern "C" void kernel_launch(void* const* d_in, const int* in_sizes, int n_in,
                              void* d_out, int out_size, void* d_ws, size_t ws_size,
                              hipStream_t stream) {
    (void)in_sizes; (void)n_in; (void)out_size; (void)ws_size;
    const float* x  = (const float*)d_in[0];   // [T,H]
    const float* rw = (const float*)d_in[1];   // [E,H]
    const float* w1 = (const float*)d_in[2];   // [E,H,I]
    const float* v1 = (const float*)d_in[3];   // [E,H,I]
    const float* w2 = (const float*)d_in[4];   // [E,I,H]
    const float* sg = (const float*)d_in[5];   // [I,H] (used transposed)
    const float* su = (const float*)d_in[6];   // [I,H] (used transposed)
    const float* sd = (const float*)d_in[7];   // [H,I] (used transposed)
    float* out = (float*)d_out;

    char*  w   = (char*)d_ws;
    size_t off = 0;
    auto carve = [&](size_t bytes) -> void* {
        void* p = w + off;
        off += (bytes + 255) & ~(size_t)255;
        return p;
    };
    unsigned short* xbf   = (unsigned short*)carve((size_t)Tn * Hn * 2); // 32MB
    unsigned short* hbuf  = (unsigned short*)carve((size_t)Tn * In * 2); // 32MB (reused)
    int*            cnts  = (int*)carve(En * sizeof(int));
    int*            basep = (int*)carve(En * sizeof(int));
    int*            eidx  = (int*)carve((size_t)Tn * sizeof(int));
    int*            eslot = (int*)carve((size_t)Tn * sizeof(int));
    float*          ewt   = (float*)carve((size_t)Tn * sizeof(float));
    int*            tlist = (int*)carve((size_t)En * Tn * sizeof(int));
    int*            tos   = (int*)carve((size_t)Tn * sizeof(int));
    float*          wslt  = (float*)carve((size_t)Tn * sizeof(float));

    zero_counts_kernel<<<1, 256, 0, stream>>>(cnts);
    convert_x_kernel<<<(int)(((size_t)Tn * Hn) / (256 * 8)), 256, 0, stream>>>(x, xbf);
    router_kernel<<<Tn * 32 / 256, 256, 0, stream>>>(x, rw, cnts, eidx, eslot, ewt, tlist);
    prefix_kernel<<<1, 1, 0, stream>>>(cnts, basep);
    build_slots_kernel<<<Tn / 256, 256, 0, stream>>>(eidx, eslot, ewt, basep, tos, wslt);

    // shared expert (dense): hbuf then plain-store to out
    dim3 g1s(In / BN, Tn / BM, 1);
    glu_stage1<false><<<g1s, 256, 0, stream>>>(xbf, sg, su, nullptr, nullptr, nullptr, hbuf);
    dim3 g2s(Hn / BN, Tn / BM, 1);
    glu_stage2<false><<<g2s, 256, 0, stream>>>(hbuf, sd, nullptr, nullptr, nullptr, nullptr, out);

    // routed experts (grouped): hbuf reused (stream-ordered), weighted += into out
    dim3 g1e(In / BN, En, Tn / BM);
    glu_stage1<true><<<g1e, 256, 0, stream>>>(xbf, w1, v1, cnts, basep, tlist, hbuf);
    dim3 g2e(Hn / BN, En, Tn / BM);
    glu_stage2<true><<<g2e, 256, 0, stream>>>(hbuf, w2, cnts, basep, tos, wslt, out);
}
